// AutoInformerEncoderLayer_19421842113004
// MI455X (gfx1250) — matrix operand order repaired and checked by reading, not compiled, not run on hardware
//
#include <hip/hip_runtime.h>
#include <hip/hip_bf16.h>

// ---------------------------------------------------------------------------
// AutoInformer encoder layer for MI455X (gfx1250, wave32, WMMA).
// Pipeline:
//   0) rowsum(Wv), sum(bv)                      (v is only used as sum over C)
//   1) transpose+f16-convert Wq,Wk,W1,W2
//   2) f16-convert x
//   3) q = x@Wq+bq, k = x@Wk+bk                 (WMMA f16 16x16x32, f32 accum)
//   4) vsum[b,t] = x[b,t,:].rowsum(Wv) + sum(bv)  (GEMV)
//   5) per (b,c) column: FFT(q + i k) -> FQ*conj(FK) -> IFFT -> softmax_t
//      -> * vsum  == attn   (all in 32KB LDS, one workgroup per column)
//   6) xmid = LN(x + attn)  (also emits f16 copy)
//   7) h1 = relu(xmid@W1+bf1)  -> f16            (WMMA)
//   8) h2 = h1@W2+bf2                            (WMMA)
//   9) out = LN(xmid + h2)
// GEMMs use double-buffered LDS with GLOBAL_LOAD_ASYNC_TO_LDS_B128 staging
// (ASYNCcnt) when the toolchain exposes the builtins; otherwise falls back to
// register staging.
// ---------------------------------------------------------------------------

typedef __attribute__((ext_vector_type(16))) _Float16 v16h;
typedef __attribute__((ext_vector_type(4)))  _Float16 v4h;
typedef __attribute__((ext_vector_type(8)))  float    v8f;
typedef __attribute__((ext_vector_type(4)))  int      v4i;

#define BB 8
#define TT 2048
#define CC 1024
#define MM (BB*TT)   // 16384 rows

// -------------------- async-to-LDS staging support -------------------------
#if __has_builtin(__builtin_amdgcn_global_load_async_to_lds_b128) && \
    __has_builtin(__builtin_amdgcn_s_wait_asynccnt)
#define ASYNC_STAGING 1
#else
#define ASYNC_STAGING 0
#endif

#if ASYNC_STAGING
typedef __attribute__((address_space(1))) v4i* gptr_b128;
typedef __attribute__((address_space(3))) v4i* lptr_b128;
__device__ __forceinline__ void async_b128(void* lds, const void* g) {
    __builtin_amdgcn_global_load_async_to_lds_b128(
        (gptr_b128)g, (lptr_b128)lds, 0, 0);
}
__device__ __forceinline__ void wait_async0() {
    __builtin_amdgcn_s_wait_asynccnt(0);
}
#endif

// ------------------------------- 0: rowsum(Wv) & sum(bv) -------------------
__global__ __launch_bounds__(256) void rowsum_kernel(const float* __restrict__ Wv,
                                                     const float* __restrict__ bv,
                                                     float* __restrict__ wvs) {
    __shared__ float red[256];
    const int tid = threadIdx.x;
    const int r   = blockIdx.x;                 // 0..1024 ; block 1024 sums bv
    const float* src = (r < CC) ? (Wv + (size_t)r * CC) : bv;
    float s = 0.f;
#pragma unroll
    for (int i = 0; i < 4; ++i) s += src[tid + i * 256];
    red[tid] = s; __syncthreads();
    for (int o = 128; o; o >>= 1) { if (tid < o) red[tid] += red[tid + o]; __syncthreads(); }
    if (tid == 0) wvs[r] = red[0];
}

// ------------------------------- 1: W -> Wt (f16, N-major) -----------------
__global__ __launch_bounds__(256) void wtrans_kernel(const float* W0, const float* W1,
                                                     const float* W2, const float* W3,
                                                     _Float16* D0, _Float16* D1,
                                                     _Float16* D2, _Float16* D3) {
    __shared__ float tile[32][33];
    const float* S; _Float16* D;
    switch (blockIdx.z) {
        case 0:  S = W0; D = D0; break;
        case 1:  S = W1; D = D1; break;
        case 2:  S = W2; D = D2; break;
        default: S = W3; D = D3; break;
    }
    const int k0 = blockIdx.y * 32, n0 = blockIdx.x * 32;
    const int tx = threadIdx.x & 31, ty = threadIdx.x >> 5;   // 32 x 8
#pragma unroll
    for (int i = 0; i < 4; ++i) {
        int r = ty + i * 8;
        tile[r][tx] = S[(size_t)(k0 + r) * CC + n0 + tx];
    }
    __syncthreads();
#pragma unroll
    for (int i = 0; i < 4; ++i) {
        int r = ty + i * 8;
        D[(size_t)(n0 + r) * CC + k0 + tx] = (_Float16)tile[tx][r];
    }
}

// ------------------------------- 2: x -> f16 -------------------------------
__global__ __launch_bounds__(256) void xcvt_kernel(const float* __restrict__ x,
                                                   _Float16* __restrict__ xh) {
    size_t i = ((size_t)blockIdx.x * 256 + threadIdx.x) * 4;
    float4 v = *(const float4*)(x + i);
    v4h o = { (_Float16)v.x, (_Float16)v.y, (_Float16)v.z, (_Float16)v.w };
    *(v4h*)(xh + i) = o;
}

// ------------------------------- 3/7/8: WMMA GEMM --------------------------
// D[M,N] = A[M,K]f16 @ Bt[N,K]f16 + bias ; optional relu ; f32 and/or f16 out.
// Block tile 128x128, K-step 32, 8 waves (2 along M x 4 along N), wave 64x32.
// Double-buffered LDS staged by async-to-LDS (ASYNCcnt) when available.
template<bool RELU, bool W32, bool W16>
__global__ __launch_bounds__(256) void gemm_f16(const _Float16* __restrict__ A,
                                                const _Float16* __restrict__ Bt,
                                                const float* __restrict__ bias,
                                                float* __restrict__ C32,
                                                _Float16* __restrict__ C16) {
    constexpr int K = 1024, N = 1024;
    constexpr int LDK = 40;                       // padded row (halves): 80B rows
    __shared__ _Float16 As[2][128 * LDK];
    __shared__ _Float16 Bs[2][128 * LDK];
    const int tid  = threadIdx.x;
    const int lane = tid & 31, wave = tid >> 5;
    const int m0 = blockIdx.y * 128, n0 = blockIdx.x * 128;
    const int wm = (wave & 1) * 64;               // wave M offset
    const int wn = (wave >> 1) * 32;              // wave N offset
    const int lr = lane & 15;                     // row/col within 16x16 tile
    const int hi = lane >> 4;                     // lane-half selector

    v8f acc[4][2] = {};

    // stage one 128x32 tile of A and Bt into LDS buffer `bufi`
    auto stage = [&](int bufi, int k0) {
#pragma unroll
        for (int i = 0; i < 2; ++i) {
            int slot = tid + i * 256;             // 512 slots of 16B per matrix
            int r = slot >> 2, ch = slot & 3;
            const _Float16* ga = A  + (size_t)(m0 + r) * K + k0 + ch * 8;
            const _Float16* gb = Bt + (size_t)(n0 + r) * K + k0 + ch * 8;
            _Float16* la = &As[bufi][r * LDK + ch * 8];
            _Float16* lb = &Bs[bufi][r * LDK + ch * 8];
#if ASYNC_STAGING
            async_b128(la, ga);
            async_b128(lb, gb);
#else
            *(uint4*)la = *(const uint4*)ga;
            *(uint4*)lb = *(const uint4*)gb;
#endif
        }
    };

    stage(0, 0);
#if ASYNC_STAGING
    wait_async0();
#endif
    __syncthreads();

    int cur = 0;
    for (int k0 = 0; k0 < K; k0 += 32) {
        if (k0 + 32 < K) stage(cur ^ 1, k0 + 32);

        // ---- build fragments (wave32 WMMA layouts) from buffer `cur`
        v16h af[4], bf[2];
#pragma unroll
        for (int i = 0; i < 4; ++i) {
            // A 16x32 f16: lane -> row M=lr ; lanes<16 K 0-7,16-23 ; >=16 K 8-15,24-31
            const _Float16* p = &As[cur][(wm + i * 16 + lr) * LDK + hi * 8];
            ((uint4*)&af[i])[0] = *(const uint4*)(p);
            ((uint4*)&af[i])[1] = *(const uint4*)(p + 16);
        }
#pragma unroll
        for (int j = 0; j < 2; ++j) {
            // B 32x16 f16: lane -> col N=lr ; lanes<16 K 0-7,16-23 ; >=16 K 8-15,24-31
            const _Float16* p = &Bs[cur][(wn + j * 16 + lr) * LDK + hi * 8];
            ((uint4*)&bf[j])[0] = *(const uint4*)(p);
            ((uint4*)&bf[j])[1] = *(const uint4*)(p + 16);
        }
#pragma unroll
        for (int i = 0; i < 4; ++i)
#pragma unroll
            for (int j = 0; j < 2; ++j)
                acc[i][j] = __builtin_amdgcn_wmma_f32_16x16x32_f16(
                    false, af[i], false, bf[j], (short)0, acc[i][j], false, false);

#if ASYNC_STAGING
        wait_async0();                            // my next-tile loads landed
#endif
        __syncthreads();                          // everyone done reading `cur`
        cur ^= 1;
    }

    // ---- epilogue: D 16x16 f32: lane col=lr ; VGPR v -> row v + 8*hi
#pragma unroll
    for (int i = 0; i < 4; ++i) {
#pragma unroll
        for (int j = 0; j < 2; ++j) {
            const int col = n0 + wn + j * 16 + lr;
            const float bvv = bias[col];
#pragma unroll
            for (int v = 0; v < 8; ++v) {
                const int row = m0 + wm + i * 16 + hi * 8 + v;
                float val = acc[i][j][v] + bvv;
                if (RELU) val = fmaxf(val, 0.f);
                if (W32) C32[(size_t)row * N + col] = val;
                if (W16) C16[(size_t)row * N + col] = (_Float16)val;
            }
        }
    }
}

// ------------------------------- 4: vsum GEMV ------------------------------
__global__ __launch_bounds__(256) void vsum_kernel(const float* __restrict__ x,
                                                   const float* __restrict__ wvs,
                                                   float* __restrict__ vsum) {
    const int wave = threadIdx.x >> 5, lane = threadIdx.x & 31;
    const int row = blockIdx.x * 8 + wave;
    const float* xr = x + (size_t)row * CC;
    float acc = 0.f;
#pragma unroll
    for (int it = 0; it < 8; ++it) {
        int k = it * 128 + lane * 4;
        float4 xa = *(const float4*)(xr + k);
        float4 wa = *(const float4*)(wvs + k);
        acc += xa.x * wa.x + xa.y * wa.y + xa.z * wa.z + xa.w * wa.w;
    }
    for (int o = 16; o; o >>= 1) acc += __shfl_down(acc, o, 32);
    if (lane == 0) vsum[row] = acc + wvs[CC];     // + sum(bv)
}

// ------------------------------- 5: FFT-corr + softmax + attn --------------
__device__ __forceinline__ float2* fft_stages(float2* src, float2* dst,
                                              float sgn, int tid) {
    int n = 2048, ls = 0;
#pragma unroll 1
    for (int st = 0; st < 11; ++st) {
        const int m = n >> 1;
        const int s = 1 << ls;
        const float thb = sgn * 6.28318530717958647692f / (float)n;
        for (int idx = tid; idx < 1024; idx += 256) {
            const int p  = idx >> ls;
            const int qq = idx & (s - 1);
            float sw, cw;
            __sincosf(thb * (float)p, &sw, &cw);
            const float2 a = src[qq + s * p];
            const float2 b = src[qq + s * (p + m)];
            const float dx = a.x - b.x, dy = a.y - b.y;
            dst[qq + s * (2 * p)]     = make_float2(a.x + b.x, a.y + b.y);
            dst[qq + s * (2 * p + 1)] = make_float2(dx * cw - dy * sw, dx * sw + dy * cw);
        }
        __syncthreads();
        float2* t = src; src = dst; dst = t;
        n >>= 1; ++ls;
    }
    return src;                                    // result buffer
}

__global__ __launch_bounds__(256) void fftcorr_kernel(const float* qb, const float* kb,
                                                      const float* __restrict__ vsum,
                                                      float* attn) {
    __shared__ float2 bufA[2048];
    __shared__ float2 bufB[2048];
    __shared__ float  red[256];
    const int tid = threadIdx.x;
    const int c   = blockIdx.x & (CC - 1);
    const int b   = blockIdx.x >> 10;
    const size_t base = (size_t)b * TT * CC + c;

    // z = q + i*k  (one complex FFT transforms both real sequences)
    for (int t = tid; t < TT; t += 256)
        bufA[t] = make_float2(qb[base + (size_t)t * CC], kb[base + (size_t)t * CC]);
    __syncthreads();

    float2* Z = fft_stages(bufA, bufB, -1.f, tid);
    float2* P = (Z == bufA) ? bufB : bufA;

    // FQ = (Z[f]+conj(Z[-f]))/2 ; FK = (Z[f]-conj(Z[-f]))/(2i) ; P = FQ*conj(FK)
    for (int f = tid; f < TT; f += 256) {
        float2 zf = Z[f];
        float2 zc = Z[(TT - f) & (TT - 1)];
        float fqr = 0.5f * (zf.x + zc.x);
        float fqi = 0.5f * (zf.y - zc.y);
        float fkr = 0.5f * (zf.y + zc.y);
        float fki = -0.5f * (zf.x - zc.x);
        P[f] = make_float2(fqr * fkr + fqi * fki, fqi * fkr - fqr * fki);
    }
    __syncthreads();

    float2* res = fft_stages(P, (P == bufA) ? bufB : bufA, +1.f, tid);

    // corr = Re(res)/N ; softmax over t ; attn = w * vsum[b,t]
    const float invN = 1.0f / (float)TT;
    float lmax = -3.4e38f;
    for (int t = tid; t < TT; t += 256) {
        float cv = res[t].x * invN;
        res[t].x = cv;
        lmax = fmaxf(lmax, cv);
    }
    red[tid] = lmax; __syncthreads();
    for (int o = 128; o; o >>= 1) { if (tid < o) red[tid] = fmaxf(red[tid], red[tid + o]); __syncthreads(); }
    const float gmax = red[0]; __syncthreads();

    float lsum = 0.f;
    for (int t = tid; t < TT; t += 256) {
        float e = __expf(res[t].x - gmax);
        res[t].y = e;
        lsum += e;
    }
    red[tid] = lsum; __syncthreads();
    for (int o = 128; o; o >>= 1) { if (tid < o) red[tid] += red[tid + o]; __syncthreads(); }
    const float inv = 1.0f / red[0];

    for (int t = tid; t < TT; t += 256)
        attn[base + (size_t)t * CC] = res[t].y * inv * vsum[b * TT + t];
}

// ------------------------------- 6/9: LayerNorm ----------------------------
__global__ __launch_bounds__(256) void ln_kernel(const float* __restrict__ a,
                                                 const float* __restrict__ bres,
                                                 const float* __restrict__ g,
                                                 const float* __restrict__ beta,
                                                 float* __restrict__ out32,
                                                 _Float16* __restrict__ out16) {
    __shared__ float red[256];
    const int row = blockIdx.x;
    const int tid = threadIdx.x;
    const size_t rb = (size_t)row * CC;
    float y[4];
    float s = 0.f;
#pragma unroll
    for (int i = 0; i < 4; ++i) {
        int c = tid + i * 256;
        y[i] = a[rb + c] + bres[rb + c];
        s += y[i];
    }
    red[tid] = s; __syncthreads();
    for (int o = 128; o; o >>= 1) { if (tid < o) red[tid] += red[tid + o]; __syncthreads(); }
    const float mu = red[0] * (1.0f / CC); __syncthreads();
    float v = 0.f;
#pragma unroll
    for (int i = 0; i < 4; ++i) { float d = y[i] - mu; v += d * d; }
    red[tid] = v; __syncthreads();
    for (int o = 128; o; o >>= 1) { if (tid < o) red[tid] += red[tid + o]; __syncthreads(); }
    const float rs = rsqrtf(red[0] * (1.0f / CC) + 1e-5f);
#pragma unroll
    for (int i = 0; i < 4; ++i) {
        int c = tid + i * 256;
        float o = (y[i] - mu) * rs * g[c] + beta[c];
        out32[rb + c] = o;
        if (out16) out16[rb + c] = (_Float16)o;
    }
}

// ------------------------------- host launch -------------------------------
extern "C" void kernel_launch(void* const* d_in, const int* in_sizes, int n_in,
                              void* d_out, int out_size, void* d_ws, size_t ws_size,
                              hipStream_t stream) {
    const float* x   = (const float*)d_in[0];
    const float* Wq  = (const float*)d_in[1];
    const float* bq  = (const float*)d_in[2];
    const float* Wk  = (const float*)d_in[3];
    const float* bk  = (const float*)d_in[4];
    const float* Wv  = (const float*)d_in[5];
    const float* bv  = (const float*)d_in[6];
    const float* g1  = (const float*)d_in[7];
    const float* be1 = (const float*)d_in[8];
    const float* W1  = (const float*)d_in[9];
    const float* bf1 = (const float*)d_in[10];
    const float* W2  = (const float*)d_in[11];
    const float* bf2 = (const float*)d_in[12];
    const float* g2  = (const float*)d_in[13];
    const float* be2 = (const float*)d_in[14];
    float* out = (float*)d_out;

    size_t off = 0;
    auto alloc = [&](size_t bytes) {
        void* p = (char*)d_ws + off;
        off += (bytes + 255) & ~(size_t)255;
        return p;
    };
    _Float16* xh    = (_Float16*)alloc((size_t)MM * CC * 2);   // 32 MB
    float*    qbuf  = (float*)   alloc((size_t)MM * CC * 4);   // 64 MB (reused as attn)
    float*    kbuf  = (float*)   alloc((size_t)MM * CC * 4);   // 64 MB (reused as h2)
    float*    xmid  = (float*)   alloc((size_t)MM * CC * 4);   // 64 MB
    _Float16* xmidh = (_Float16*)alloc((size_t)MM * CC * 2);   // 32 MB
    _Float16* h1h   = (_Float16*)alloc((size_t)MM * CC * 2);   // 32 MB
    _Float16* wqt   = (_Float16*)alloc((size_t)CC * CC * 2);
    _Float16* wkt   = (_Float16*)alloc((size_t)CC * CC * 2);
    _Float16* w1t   = (_Float16*)alloc((size_t)CC * CC * 2);
    _Float16* w2t   = (_Float16*)alloc((size_t)CC * CC * 2);
    float*    wvs   = (float*)   alloc((CC + 32) * 4);         // rowsums + sum(bv)
    float*    vsumb = (float*)   alloc((size_t)MM * 4);

    const dim3 gGemm(CC / 128, MM / 128);                      // (8, 128)

    rowsum_kernel<<<CC + 1, 256, 0, stream>>>(Wv, bv, wvs);
    wtrans_kernel<<<dim3(32, 32, 4), 256, 0, stream>>>(Wq, Wk, W1, W2, wqt, wkt, w1t, w2t);
    xcvt_kernel<<<(MM * CC) / 1024, 256, 0, stream>>>(x, xh);

    gemm_f16<false, true, false><<<gGemm, 256, 0, stream>>>(xh, wqt, bq, qbuf, nullptr);
    gemm_f16<false, true, false><<<gGemm, 256, 0, stream>>>(xh, wkt, bk, kbuf, nullptr);
    vsum_kernel<<<MM / 8, 256, 0, stream>>>(x, wvs, vsumb);

    fftcorr_kernel<<<BB * CC, 256, 0, stream>>>(qbuf, kbuf, vsumb, qbuf /*attn in place*/);

    ln_kernel<<<MM, 256, 0, stream>>>(x, qbuf, g1, be1, xmid, xmidh);

    gemm_f16<true, false, true><<<gGemm, 256, 0, stream>>>(xmidh, w1t, bf1, nullptr, h1h);
    gemm_f16<false, true, false><<<gGemm, 256, 0, stream>>>(h1h, w2t, bf2, kbuf /*h2*/, nullptr);

    ln_kernel<<<MM, 256, 0, stream>>>(xmid, kbuf, g2, be2, out, nullptr);
}
